// Pointnet2Backbone_41008347742444
// MI455X (gfx1250) — matrix operand order repaired and checked by reading, not compiled, not run on hardware
//
#include <hip/hip_runtime.h>
#include <math.h>

#define BATCH 2
#define NRAW  20000
#define NHEAD 8

typedef __attribute__((ext_vector_type(16))) __bf16          bf16x16;
typedef __attribute__((ext_vector_type(16))) unsigned short  u16x16;
typedef __attribute__((ext_vector_type(8)))  float           f32x8;

static __device__ __forceinline__ unsigned short f32_to_bf16(float f) {
  union { float f; unsigned u; } c; c.f = f;
  unsigned u = c.u + 0x7FFFu + ((c.u >> 16) & 1u);   // round-to-nearest-even
  return (unsigned short)(u >> 16);
}

// ---------------------------------------------------------------------------
// Tensor Data Mover: 2D f32 tile (rows x 32 cols, row stride = stride_elems)
// from global into LDS at byte offset ldsOff. tensor_dim0 = cols_avail gives
// hardware zero-fill past the K edge; LDS padding (4 dw per 32 dw) yields a
// 36-dword LDS row stride for bank-conflict-free fragment reads.
// ---------------------------------------------------------------------------
#if __has_builtin(__builtin_amdgcn_tensor_load_to_lds)
#define HAVE_TDM 1
typedef __attribute__((ext_vector_type(4))) unsigned int u32x4;
typedef __attribute__((ext_vector_type(8))) int          i32x8;
typedef __attribute__((ext_vector_type(4))) int          i32x4;

static __device__ __forceinline__ void tdm_load_tile_f32(
    const float* gsrc, unsigned ldsOff, int cols_avail, int rows, int stride_elems)
{
  const unsigned long long ga = (unsigned long long)(const void*)gsrc;
  const unsigned long long st = (unsigned long long)stride_elems;
  u32x4 g0;
  g0[0] = 1u;                                                  // count=1, user D#
  g0[1] = ldsOff;                                              // lds_addr (bytes)
  g0[2] = (unsigned)(ga & 0xFFFFFFFFull);                      // global_addr[31:0]
  g0[3] = (unsigned)((ga >> 32) & 0x01FFFFFFull) | (2u << 30); // [56:32] | type=2
  const unsigned td0 = (unsigned)cols_avail;
  const unsigned td1 = (unsigned)rows;
  i32x8 g1;
  g1[0] = (int)((2u << 16) | (1u << 20) | (4u << 22) | (3u << 25));
          // data_size=4B | pad_enable | pad_interval: 32 dw | pad_amount: 4 dw
  g1[1] = (int)((td0 & 0xFFFFu) << 16);                        // tensor_dim0[15:0]
  g1[2] = (int)(((td0 >> 16) & 0xFFFFu) | ((td1 & 0xFFFFu) << 16));
  g1[3] = (int)(((td1 >> 16) & 0xFFFFu) | (32u << 16));        // tile_dim0 = 32
  g1[4] = (int)(td1 & 0xFFFFu);                                // tile_dim1 = rows
  g1[5] = (int)(st & 0xFFFFFFFFull);                           // dim0 stride lo
  g1[6] = (int)((st >> 32) & 0xFFFFull);                       // dim0 stride hi
  g1[7] = 0;
  i32x4 z4 = {0, 0, 0, 0};
#if __has_include(<hip/amd_detail/amd_gfx1250_TDM.h>)
  i32x8 z8 = {0, 0, 0, 0, 0, 0, 0, 0};
  __builtin_amdgcn_tensor_load_to_lds(g0, g1, z4, z4, z8, 0);  // clang-23 6-arg form
#else
  __builtin_amdgcn_tensor_load_to_lds(g0, g1, z4, z4, 0);      // ROCm 7.2 5-arg form
#endif
}
#endif

// ---------------------------------------------------------------------------
// Fused GEMM via V_WMMA_F32_16X16X32_BF16:
//   Y[M,N] = act((X[M,K] @ W[N,K]^T + bias) [* gamma + beta]),  act=ReLU if RELU
// Requires M % 128 == 0 and N % 16 == 0 (true for every layer in this net),
// so the only ragged dimension is K, which TDM zero-fills. A/B tiles are
// staged to LDS by the TDM (double buffered, issued by wave 0), fragments are
// read as contiguous 16B-aligned chunks and converted f32->bf16 in-register.
// ---------------------------------------------------------------------------
#define LDA 36   // padded LDS row stride in f32 (32 data + 4 pad dwords)

template <int RELU, int BN>
__global__ __launch_bounds__(256) void gemm_wmma_kernel(
    const float* __restrict__ X, const float* __restrict__ W,
    const float* __restrict__ bias, const float* __restrict__ gamma,
    const float* __restrict__ beta, float* __restrict__ Y,
    int M, int N, int K)
{
  __shared__ float lA[2][128 * LDA];
  __shared__ float lB[2][16 * LDA];

  const int lane    = threadIdx.x & 31;
  const int wave    = threadIdx.x >> 5;
  const int half    = lane >> 4;
  const int l16     = lane & 15;
  const int rowBase = blockIdx.x * 128;
  const int colBase = blockIdx.y * 16;
  const int nSteps  = (K + 31) >> 5;
  (void)M;

#if defined(HAVE_TDM)
  if (wave == 0) {   // wave-uniform branch; TDM is a per-wave DMA op
    tdm_load_tile_f32(X + (size_t)rowBase * K, (unsigned)(size_t)(void*)&lA[0][0], K, 128, K);
    tdm_load_tile_f32(W + (size_t)colBase * K, (unsigned)(size_t)(void*)&lB[0][0], K, 16,  K);
  }
#endif

  f32x8 acc = {};
  for (int s = 0; s < nSteps; ++s) {
    const int k0  = s << 5;
    const int buf = s & 1;
#if defined(HAVE_TDM)
    if (wave == 0) {
      if (s + 1 < nSteps) {                      // prefetch next step into buf^1
        const int k1 = k0 + 32;
        tdm_load_tile_f32(X + (size_t)rowBase * K + k1,
                          (unsigned)(size_t)(void*)&lA[buf ^ 1][0], K - k1, 128, K);
        tdm_load_tile_f32(W + (size_t)colBase * K + k1,
                          (unsigned)(size_t)(void*)&lB[buf ^ 1][0], K - k1, 16,  K);
        __builtin_amdgcn_s_wait_tensorcnt(2);    // in-order: step s's 2 loads done
      } else {
        __builtin_amdgcn_s_wait_tensorcnt(0);
      }
    }
    __syncthreads();
#else
    // Branch-free cooperative staging fallback (clamp-load + select-zero).
    for (int i = 0; i < 16; ++i) {
      const int lin = (int)threadIdx.x + (i << 8);
      const int r = lin >> 5, c = lin & 31, k = k0 + c;
      const int kc = k < K ? k : K - 1;
      const float v = X[(size_t)(rowBase + r) * K + kc];
      lA[buf][r * LDA + c] = (k < K) ? v : 0.0f;
    }
    for (int i = 0; i < 2; ++i) {
      const int lin = (int)threadIdx.x + (i << 8);
      const int r = lin >> 5, c = lin & 31, k = k0 + c;
      const int kc = k < K ? k : K - 1;
      const float v = W[(size_t)(colBase + r) * K + kc];
      lB[buf][r * LDA + c] = (k < K) ? v : 0.0f;
    }
    __syncthreads();
#endif
    // Fragment assembly (contiguous k = half*16 + e in both A and B), contiguous 16B-aligned reads.
    const float* Ar = &lA[buf][(wave * 16 + l16) * LDA];
    const float* Br = &lB[buf][l16 * LDA];
    u16x16 au, bu;
#pragma unroll
    for (int e = 0; e < 16; ++e)
      au[e] = f32_to_bf16(Ar[half * 16 + e]);           // K = half*16 + e
#pragma unroll
    for (int e = 0; e < 16; ++e)
      bu[e] = f32_to_bf16(Br[half * 16 + e]);           // K = half*16 + e
    acc = __builtin_amdgcn_wmma_f32_16x16x32_bf16(
        false, __builtin_bit_cast(bf16x16, au),
        false, __builtin_bit_cast(bf16x16, bu),
        (short)0, acc, false, false);
    __syncthreads();
  }

  const int   nCol = colBase + l16;
  const float bb = bias[nCol];
  const float gg = BN ? gamma[nCol] : 1.0f;
  const float be = BN ? beta[nCol]  : 0.0f;
#pragma unroll
  for (int v = 0; v < 8; ++v) {                  // C/D layout: row = v + 8*half
    const int row = rowBase + wave * 16 + v + 8 * half;
    float y = acc[v] + bb;
    if (BN)   y = y * gg + be;
    if (RELU) y = fmaxf(y, 0.0f);
    Y[(size_t)row * N + nCol] = y;
  }
}

// ----------------------------------------------------------------- FPS
__global__ __launch_bounds__(1024) void fps_kernel(
    const float* __restrict__ xyz, float* __restrict__ dist,
    int* __restrict__ inds, int n, int npoint)
{
  const int b = blockIdx.x;
  const float* p  = xyz + (size_t)b * n * 3;
  float* dst = dist + (size_t)b * n;
  int*   out = inds + (size_t)b * npoint;
  __shared__ float sval[1024];
  __shared__ int   sidx[1024];
  __shared__ float cx, cy, cz;
  const int t = threadIdx.x;
  for (int j = t; j < n; j += 1024) dst[j] = 1e10f;
  int far = 0;
  if (t == 0) { cx = p[0]; cy = p[1]; cz = p[2]; }
  __syncthreads();
  for (int i = 0; i < npoint; ++i) {
    if (t == 0) out[i] = far;
    const float fx = cx, fy = cy, fz = cz;
    float bestv = -1.0f; int besti = 0;
    for (int j = t; j < n; j += 1024) {
      const float dx = p[j*3+0]-fx, dy = p[j*3+1]-fy, dz = p[j*3+2]-fz;
      float m = fminf(dst[j], dx*dx + dy*dy + dz*dz);
      dst[j] = m;
      if (m > bestv) { bestv = m; besti = j; }
    }
    sval[t] = bestv; sidx[t] = besti;
    __syncthreads();
    for (int s = 512; s > 0; s >>= 1) {
      if (t < s && sval[t + s] > sval[t]) { sval[t] = sval[t+s]; sidx[t] = sidx[t+s]; }
      __syncthreads();
    }
    far = sidx[0];
    if (t == 0) { cx = p[far*3]; cy = p[far*3+1]; cz = p[far*3+2]; }
    __syncthreads();
  }
}

// ----------------------------------------------------------------- ball query
__global__ void ball_query_kernel(const float* __restrict__ xyz,
                                  const float* __restrict__ new_xyz,
                                  int* __restrict__ idx,
                                  int n, int np, int nsample, float r2)
{
  const int wpb  = blockDim.x >> 5;
  const int q    = blockIdx.x * wpb + (threadIdx.x >> 5);
  const int lane = threadIdx.x & 31;
  if (q >= BATCH * np) return;
  const int b = q / np;
  const float* P = xyz + (size_t)b * n * 3;
  const float* C = new_xyz + (size_t)q * 3;
  int* out = idx + (size_t)q * nsample;
  const float cx = C[0], cy = C[1], cz = C[2];
  int cnt = 0, f0 = -1;
  for (int j0 = 0; j0 < n && cnt < nsample; j0 += 32) {
    const int j = j0 + lane;
    bool pass = false;
    if (j < n) {
      const float dx = P[j*3]-cx, dy = P[j*3+1]-cy, dz = P[j*3+2]-cz;
      pass = (dx*dx + dy*dy + dz*dz) < r2;
    }
    const unsigned mask = __builtin_amdgcn_ballot_w32(pass);   // wave32 ballot
    if (f0 < 0 && mask) f0 = j0 + (__ffs(mask) - 1);
    const int off = cnt + __popc(mask & ((1u << lane) - 1u));
    if (pass && off < nsample) out[off] = j;
    cnt += __popc(mask);
  }
  if (cnt > nsample) cnt = nsample;
  if (f0 < 0) f0 = 0;
  for (int s = cnt + lane; s < nsample; s += 32) out[s] = f0;
}

// ----------------------------------------------------------------- gather/group
__global__ void gather_points_kernel(const float* __restrict__ src,
                                     const int* __restrict__ inds,
                                     float* __restrict__ dst, int n, int np, int C)
{
  const int gid = blockIdx.x * blockDim.x + threadIdx.x;
  if (gid >= BATCH * np * C) return;
  const int c = gid % C, p = (gid / C) % np, b = gid / (C * np);
  const int j = inds[b * np + p];
  dst[gid] = src[((size_t)b * n + j) * C + c];
}

__global__ void group_kernel(const float* __restrict__ xyz,
                             const float* __restrict__ feats,   // may be null
                             const float* __restrict__ new_xyz,
                             const int*   __restrict__ idx,
                             float* __restrict__ g_xyz,         // (B,np,ns,3)
                             float* __restrict__ g_in,          // (B,np,ns,3+Cf)
                             int n, int np, int ns, int Cf, float inv_r)
{
  const int gid = blockIdx.x * blockDim.x + threadIdx.x;
  if (gid >= BATCH * np * ns) return;
  const int p = (gid / ns) % np, b = gid / (ns * np);
  const int j = idx[gid];
  const float* P = xyz + ((size_t)b * n + j) * 3;
  const float* C = new_xyz + ((size_t)b * np + p) * 3;
  const float gx = (P[0]-C[0]) * inv_r, gy = (P[1]-C[1]) * inv_r, gz = (P[2]-C[2]) * inv_r;
  g_xyz[(size_t)gid*3+0] = gx; g_xyz[(size_t)gid*3+1] = gy; g_xyz[(size_t)gid*3+2] = gz;
  float* o = g_in + (size_t)gid * (3 + Cf);
  o[0] = gx; o[1] = gy; o[2] = gz;
  if (feats) {
    const float* F = feats + ((size_t)b * n + j) * Cf;
    for (int c = 0; c < Cf; ++c) o[3 + c] = F[c];
  }
}

__global__ void maxpool_kernel(const float* __restrict__ g, float* __restrict__ out,
                               int np, int ns, int C)
{
  const int gid = blockIdx.x * blockDim.x + threadIdx.x;
  if (gid >= BATCH * np * C) return;
  const int c = gid % C, p = (gid / C) % np, b = gid / (C * np);
  const size_t base = (((size_t)b * np + p) * ns) * C + c;
  float m = -1e30f;
  for (int s = 0; s < ns; ++s) m = fmaxf(m, g[base + (size_t)s * C]);
  out[gid] = m;
}

// ----------------------------------------------------------------- layernorm/add
__global__ void layer_norm_kernel(const float* __restrict__ x, float* __restrict__ y,
                                  const float* __restrict__ g, const float* __restrict__ be,
                                  int M, int d)
{
  const int row  = blockIdx.x * (blockDim.x >> 5) + (threadIdx.x >> 5);
  const int lane = threadIdx.x & 31;
  if (row >= M) return;
  const float* xr = x + (size_t)row * d;
  float s = 0.f, ss = 0.f;
  for (int c = lane; c < d; c += 32) { const float v = xr[c]; s += v; ss += v * v; }
  for (int o = 16; o > 0; o >>= 1) { s += __shfl_xor(s, o, 32); ss += __shfl_xor(ss, o, 32); }
  const float mean = s / d;
  const float inv  = rsqrtf(ss / d - mean * mean + 1e-5f);
  float* yr = y + (size_t)row * d;
  for (int c = lane; c < d; c += 32) yr[c] = (xr[c] - mean) * inv * g[c] + be[c];
}

__global__ void add_inplace_kernel(float* __restrict__ dst, const float* __restrict__ src,
                                   size_t n)
{
  const size_t i = (size_t)blockIdx.x * blockDim.x + threadIdx.x;
  if (i < n) dst[i] += src[i];
}

// ----------------------------------------------------------------- attention core
__global__ void attn_core_kernel(const float* __restrict__ q,   // (M1,d)
                                 const float* __restrict__ k,   // (M1,ns,d)
                                 const float* __restrict__ v,   // (M1,ns,d)
                                 float* __restrict__ out,       // (M1,d)
                                 int M1, int ns, int d)
{
  const int wpb  = blockDim.x >> 5;
  const int gw   = blockIdx.x * wpb + (threadIdx.x >> 5);
  const int lane = threadIdx.x & 31;
  if (gw >= M1 * NHEAD) return;
  const int h = gw % NHEAD, p = gw / NHEAD, dh = d / NHEAD;
  const float* qp = q + (size_t)p * d + h * dh;
  const float* kp = k + (size_t)p * ns * d + h * dh;
  const float* vp = v + (size_t)p * ns * d + h * dh;
  const float scale = rsqrtf((float)dh);
  float sc0 = -1e30f, sc1 = -1e30f;
  if (lane < ns) {
    float a = 0.f;
    for (int e = 0; e < dh; ++e) a += qp[e] * kp[(size_t)lane * d + e];
    sc0 = a * scale;
  }
  if (lane + 32 < ns) {
    float a = 0.f;
    for (int e = 0; e < dh; ++e) a += qp[e] * kp[(size_t)(lane + 32) * d + e];
    sc1 = a * scale;
  }
  float mx = fmaxf(sc0, sc1);
  for (int o = 16; o > 0; o >>= 1) mx = fmaxf(mx, __shfl_xor(mx, o, 32));
  const float e0 = (lane < ns)      ? __expf(sc0 - mx) : 0.f;
  const float e1 = (lane + 32 < ns) ? __expf(sc1 - mx) : 0.f;
  float sum = e0 + e1;
  for (int o = 16; o > 0; o >>= 1) sum += __shfl_xor(sum, o, 32);
  const float inv = 1.f / sum;
  const float w0 = e0 * inv, w1 = e1 * inv;
  float* op = out + (size_t)p * d + h * dh;
  for (int e = 0; e < dh; ++e) {
    float a = 0.f;
    if (lane < ns)      a += w0 * vp[(size_t)lane * d + e];
    if (lane + 32 < ns) a += w1 * vp[(size_t)(lane + 32) * d + e];
    for (int o = 16; o > 0; o >>= 1) a += __shfl_xor(a, o, 32);
    if (lane == 0) op[e] = a;
  }
}

// ----------------------------------------------------------------- FP module
__global__ void knn3_kernel(const float* __restrict__ unk, const float* __restrict__ kn,
                            int* __restrict__ idx3, float* __restrict__ w3,
                            int nu, int nk)
{
  const int gid = blockIdx.x * blockDim.x + threadIdx.x;
  if (gid >= BATCH * nu) return;
  const int b = gid / nu;
  const float* U = unk + (size_t)gid * 3;
  const float* K = kn + (size_t)b * nk * 3;
  float d0 = 1e30f, d1 = 1e30f, d2 = 1e30f; int i0 = 0, i1 = 0, i2 = 0;
  const float ux = U[0], uy = U[1], uz = U[2];
  for (int j = 0; j < nk; ++j) {
    const float dx = K[j*3]-ux, dy = K[j*3+1]-uy, dz = K[j*3+2]-uz;
    const float dd = dx*dx + dy*dy + dz*dz;
    if (dd < d0)      { d2 = d1; i2 = i1; d1 = d0; i1 = i0; d0 = dd; i0 = j; }
    else if (dd < d1) { d2 = d1; i2 = i1; d1 = dd; i1 = j; }
    else if (dd < d2) { d2 = dd; i2 = j; }
  }
  float w0 = 1.f/(d0+1e-8f), w1 = 1.f/(d1+1e-8f), w2 = 1.f/(d2+1e-8f);
  const float wn = 1.f / (w0 + w1 + w2);
  idx3[gid*3+0] = i0; idx3[gid*3+1] = i1; idx3[gid*3+2] = i2;
  w3[gid*3+0] = w0*wn; w3[gid*3+1] = w1*wn; w3[gid*3+2] = w2*wn;
}

__global__ void interp_concat_kernel(const float* __restrict__ kn_feat,  // (B,nk,C)
                                     const float* __restrict__ unk_feat, // (B,nu,C)
                                     const int* __restrict__ idx3, const float* __restrict__ w3,
                                     float* __restrict__ xout,           // (B,nu,2C)
                                     int nu, int nk, int C)
{
  const int gid = blockIdx.x * blockDim.x + threadIdx.x;
  if (gid >= BATCH * nu * C) return;
  const int c = gid % C, p = (gid / C) % nu, b = gid / (C * nu);
  const int row = b * nu + p;
  float acc = 0.f;
  for (int t = 0; t < 3; ++t)
    acc += w3[row*3+t] * kn_feat[((size_t)b * nk + idx3[row*3+t]) * C + c];
  float* o = xout + (size_t)row * (2 * C);
  o[c]     = acc;
  o[C + c] = unk_feat[(size_t)row * C + c];
}

__global__ void transpose_kernel(const float* __restrict__ x, float* __restrict__ y,
                                 int np, int C)   // (B,np,C) -> (B,C,np)
{
  const int gid = blockIdx.x * blockDim.x + threadIdx.x;
  if (gid >= BATCH * np * C) return;
  const int c = gid % C, p = (gid / C) % np, b = gid / (C * np);
  y[((size_t)b * C + c) * np + p] = x[gid];
}

// ===========================================================================
// Host orchestration
// ===========================================================================
struct Dense { const float *W, *b; };
struct BNp   { const float *be, *g; };
struct Attn  { const float *Wi, *Wo, *bi, *bo; };
struct Pos   { BNp bn; Dense l1, l2; };
struct Dec   { Attn ca; Pos cross; Dense lin1, lin2; BNp n1, n2, n3, nmem; Attn sa; Pos self; };
struct MlpLayer { const float *W, *b, *be, *g; };

static inline int cdiv(long long a, long long b) { return (int)((a + b - 1) / b); }

static void gemm_launch(hipStream_t st, const float* X, const float* W, const float* b,
                        const float* g, const float* be, float* Y,
                        int M, int N, int K, bool relu, bool bn)
{
  dim3 grid(cdiv(M, 128), cdiv(N, 16));
  if (relu && bn) gemm_wmma_kernel<1,1><<<grid, 256, 0, st>>>(X, W, b, g, be, Y, M, N, K);
  else if (relu)  gemm_wmma_kernel<1,0><<<grid, 256, 0, st>>>(X, W, b, g, be, Y, M, N, K);
  else            gemm_wmma_kernel<0,0><<<grid, 256, 0, st>>>(X, W, b, g, be, Y, M, N, K);
}

static void run_dec(hipStream_t st, const Dec& P, const float* xyz_c, float* feat,
                    const float* gxyz, const float* gfeat,
                    int M1, int ns, int d, int ff,
                    float* KEY, float* KB, float* VB, float* S0,
                    float* SP0, float* SP1, float* SP2)
{
  const size_t M2 = (size_t)M1 * ns;
  const size_t dd = (size_t)d * d;
  // query = feat + pos_embed(xyz_c, self_pos)
  gemm_launch(st, xyz_c, P.self.l1.W, P.self.l1.b, P.self.bn.g, P.self.bn.be, SP0, M1, d, 3, true, true);
  gemm_launch(st, SP0, P.self.l2.W, P.self.l2.b, 0, 0, SP1, M1, d, d, false, false);
  add_inplace_kernel<<<cdiv((size_t)M1*d, 256), 256, 0, st>>>(feat, SP1, (size_t)M1 * d);
  // key = gfeat + pos_embed(gxyz, cross_pos)
  gemm_launch(st, gxyz, P.cross.l1.W, P.cross.l1.b, P.cross.bn.g, P.cross.bn.be, S0, (int)M2, d, 3, true, true);
  gemm_launch(st, S0, P.cross.l2.W, P.cross.l2.b, 0, 0, KEY, (int)M2, d, d, false, false);
  add_inplace_kernel<<<cdiv(M2 * d, 256), 256, 0, st>>>(KEY, gfeat, M2 * d);
  // tgt = query + self_attn1(LN(query))
  layer_norm_kernel<<<cdiv(M1, 8), 256, 0, st>>>(feat, SP0, P.n1.g, P.n1.be, M1, d);
  gemm_launch(st, SP0, P.sa.Wi + 2*dd, P.sa.bi + 2*d, 0, 0, SP1, M1, d, d, false, false);
  gemm_launch(st, SP1, P.sa.Wo, P.sa.bo, 0, 0, SP2, M1, d, d, false, false);
  add_inplace_kernel<<<cdiv((size_t)M1*d, 256), 256, 0, st>>>(feat, SP2, (size_t)M1 * d);
  // cross attention
  layer_norm_kernel<<<cdiv(M1, 8), 256, 0, st>>>(feat, SP0, P.n2.g, P.n2.be, M1, d);
  layer_norm_kernel<<<cdiv(M2, 8), 256, 0, st>>>(KEY, KEY, P.nmem.g, P.nmem.be, (int)M2, d);
  gemm_launch(st, SP0, P.ca.Wi,        P.ca.bi,       0, 0, SP1, M1,      d, d, false, false); // q
  gemm_launch(st, KEY, P.ca.Wi + dd,   P.ca.bi + d,   0, 0, KB,  (int)M2, d, d, false, false); // k
  gemm_launch(st, KEY, P.ca.Wi + 2*dd, P.ca.bi + 2*d, 0, 0, VB,  (int)M2, d, d, false, false); // v
  attn_core_kernel<<<cdiv((size_t)M1 * NHEAD, 4), 128, 0, st>>>(SP1, KB, VB, SP2, M1, ns, d);
  gemm_launch(st, SP2, P.ca.Wo, P.ca.bo, 0, 0, SP0, M1, d, d, false, false);
  add_inplace_kernel<<<cdiv((size_t)M1*d, 256), 256, 0, st>>>(feat, SP0, (size_t)M1 * d);
  // FFN
  layer_norm_kernel<<<cdiv(M1, 8), 256, 0, st>>>(feat, SP0, P.n3.g, P.n3.be, M1, d);
  gemm_launch(st, SP0, P.lin1.W, P.lin1.b, 0, 0, SP1, M1, ff, d, true, false);
  gemm_launch(st, SP1, P.lin2.W, P.lin2.b, 0, 0, SP2, M1, d, ff, false, false);
  add_inplace_kernel<<<cdiv((size_t)M1*d, 256), 256, 0, st>>>(feat, SP2, (size_t)M1 * d);
}

static void run_sa(hipStream_t st, const float* xyz_in, int n_in, const float* feat_in, int Cf,
                   int np, int ns, float radius, const MlpLayer* L,
                   int C1, int C2, int C3,
                   float* new_xyz, float* new_feat, float* g_xyz, float* GF,
                   int* inds, int* idx, float* distws, float* S0, float* S1)
{
  fps_kernel<<<BATCH, 1024, 0, st>>>(xyz_in, distws, inds, n_in, np);
  gather_points_kernel<<<cdiv((size_t)BATCH*np*3, 256), 256, 0, st>>>(xyz_in, inds, new_xyz, n_in, np, 3);
  ball_query_kernel<<<cdiv((size_t)BATCH*np, 4), 128, 0, st>>>(xyz_in, new_xyz, idx, n_in, np, ns, radius*radius);
  const int M2 = BATCH * np * ns;
  group_kernel<<<cdiv(M2, 256), 256, 0, st>>>(xyz_in, feat_in, new_xyz, idx, g_xyz, S0, n_in, np, ns, Cf, 1.0f/radius);
  gemm_launch(st, S0, L[0].W, L[0].b, L[0].g, L[0].be, S1, M2, C1, 3 + Cf, true, true);
  gemm_launch(st, S1, L[1].W, L[1].b, L[1].g, L[1].be, S0, M2, C2, C1, true, true);
  gemm_launch(st, S0, L[2].W, L[2].b, L[2].g, L[2].be, GF, M2, C3, C2, true, true);
  maxpool_kernel<<<cdiv((size_t)BATCH*np*C3, 256), 256, 0, st>>>(GF, new_feat, np, ns, C3);
}

static void run_fp(hipStream_t st, const float* ux, const float* kx,
                   const float* uf, const float* kf, int nu, int nk, int C,
                   const MlpLayer* L, int Cmid, int Cout, float* out,
                   int* idx3, float* w3, float* S0, float* S1)
{
  knn3_kernel<<<cdiv((size_t)BATCH*nu, 256), 256, 0, st>>>(ux, kx, idx3, w3, nu, nk);
  interp_concat_kernel<<<cdiv((size_t)BATCH*nu*C, 256), 256, 0, st>>>(kf, uf, idx3, w3, S0, nu, nk, C);
  gemm_launch(st, S0, L[0].W, L[0].b, L[0].g, L[0].be, S1, BATCH*nu, Cmid, 2*C, true, true);
  gemm_launch(st, S1, L[1].W, L[1].b, L[1].g, L[1].be, out, BATCH*nu, Cout, Cmid, true, true);
}

extern "C" void kernel_launch(void* const* d_in, const int* in_sizes, int n_in,
                              void* d_out, int out_size, void* d_ws, size_t ws_size,
                              hipStream_t stream)
{
  (void)out_size; (void)ws_size;
  // ---- input mapping: pointcloud is either first (dict insertion order) or
  // last (fully sorted flatten). Params follow JAX tree-flatten sorted-key order.
  const bool pcFirst = (n_in > 0 && in_sizes[0] == BATCH * NRAW * 3);
  const float* pc = (const float*)d_in[pcFirst ? 0 : n_in - 1];
  int cur = pcFirst ? 1 : 0;
  auto F     = [&]() { return (const float*)d_in[cur++]; };
  auto dense = [&]() { Dense x; x.W = F(); x.b = F(); return x; };          // W, b
  auto bnp   = [&]() { BNp x; x.be = F(); x.g = F(); return x; };           // be, g
  auto attn  = [&]() { Attn a; a.Wi = F(); a.Wo = F(); a.bi = F(); a.bo = F(); return a; };
  auto pos   = [&]() { Pos p; p.bn = bnp(); p.l1 = dense(); p.l2 = dense(); return p; };
  auto dec   = [&]() { Dec d; d.ca = attn(); d.cross = pos(); d.lin1 = dense(); d.lin2 = dense();
                       d.n1 = bnp(); d.n2 = bnp(); d.n3 = bnp(); d.nmem = bnp();
                       d.sa = attn(); d.self = pos(); return d; };
  auto mlpl  = [&]() { MlpLayer l; l.W = F(); l.b = F(); l.be = F(); l.g = F(); return l; };

  // sorted param keys: dec1, dec2, dec4, fp1, fp2, sa1, sa2, sa3, sa4
  Dec dec1 = dec(), dec2 = dec(), dec4 = dec();
  MlpLayer fp1L[2] = { mlpl(), mlpl() };
  MlpLayer fp2L[2] = { mlpl(), mlpl() };
  MlpLayer sa1L[3] = { mlpl(), mlpl(), mlpl() };
  MlpLayer sa2L[3] = { mlpl(), mlpl(), mlpl() };
  MlpLayer sa3L[3] = { mlpl(), mlpl(), mlpl() };
  MlpLayer sa4L[3] = { mlpl(), mlpl(), mlpl() };

  // ---- workspace bump allocator
  size_t off = 0;
  auto allocf = [&](size_t nf) { float* p = (float*)((char*)d_ws + off);
                                 off += ((nf * 4 + 255) & ~(size_t)255); return p; };
  auto alloci = [&](size_t ni) { int* p = (int*)((char*)d_ws + off);
                                 off += ((ni * 4 + 255) & ~(size_t)255); return p; };

  const size_t M2_1 = (size_t)BATCH * 2048 * 64;            // 262144 (largest M2)
  float* S0  = allocf(M2_1 * 128);
  float* S1  = allocf(M2_1 * 128);
  float* KEY = allocf(M2_1 * 128);
  float* KB  = allocf(M2_1 * 128);
  float* VB  = allocf(M2_1 * 128);
  float* GF  = allocf(M2_1 * 128);
  float* GX  = allocf(M2_1 * 3);
  float* SP0 = allocf((size_t)BATCH * 2048 * 512);
  float* SP1 = allocf((size_t)BATCH * 2048 * 512);
  float* SP2 = allocf((size_t)BATCH * 2048 * 512);
  float* distws = allocf((size_t)BATCH * NRAW);
  int*   inds = alloci((size_t)BATCH * 2048);
  int*   idx  = alloci(M2_1);
  int*   idx3 = alloci((size_t)BATCH * 1024 * 3);
  float* w3   = allocf((size_t)BATCH * 1024 * 3);
  float* x1 = allocf((size_t)BATCH*2048*3); float* f1 = allocf((size_t)BATCH*2048*128);
  float* x2 = allocf((size_t)BATCH*1024*3); float* f2 = allocf((size_t)BATCH*1024*256);
  float* x3 = allocf((size_t)BATCH*512*3);  float* f3 = allocf((size_t)BATCH*512*256);
  float* x4 = allocf((size_t)BATCH*256*3);  float* f4 = allocf((size_t)BATCH*256*256);
  float* fp1o = allocf((size_t)BATCH*512*256);
  float* fp2o = allocf((size_t)BATCH*1024*288);

  // ---- pipeline
  run_sa(stream, pc, NRAW, nullptr, 0, 2048, 64, 0.2f, sa1L, 64, 64, 128,
         x1, f1, GX, GF, inds, idx, distws, S0, S1);
  run_dec(stream, dec1, x1, f1, GX, GF, BATCH*2048, 64, 128, 256, KEY, KB, VB, S0, SP0, SP1, SP2);

  run_sa(stream, x1, 2048, f1, 128, 1024, 32, 0.4f, sa2L, 128, 128, 256,
         x2, f2, GX, GF, inds, idx, distws, S0, S1);
  run_dec(stream, dec2, x2, f2, GX, GF, BATCH*1024, 32, 256, 512, KEY, KB, VB, S0, SP0, SP1, SP2);

  run_sa(stream, x2, 1024, f2, 256, 512, 16, 0.8f, sa3L, 128, 128, 256,
         x3, f3, GX, GF, inds, idx, distws, S0, S1);
  run_dec(stream, dec4, x3, f3, GX, GF, BATCH*512, 16, 256, 512, KEY, KB, VB, S0, SP0, SP1, SP2);

  run_sa(stream, x3, 512, f3, 256, 256, 16, 1.2f, sa4L, 128, 128, 256,
         x4, f4, GX, GF, inds, idx, distws, S0, S1);
  run_dec(stream, dec4, x4, f4, GX, GF, BATCH*256, 16, 256, 512, KEY, KB, VB, S0, SP0, SP1, SP2);

  run_fp(stream, x3, x4, f3, f4, 512, 256, 256, fp1L, 256, 256, fp1o, idx3, w3, S0, S1);
  run_fp(stream, x2, x3, f2, fp1o, 1024, 512, 256, fp2L, 256, 288, fp2o, idx3, w3, S0, S1);

  transpose_kernel<<<cdiv((size_t)BATCH*1024*288, 256), 256, 0, stream>>>(fp2o, (float*)d_out, 1024, 288);
}
